// SumKernel_83605833384118
// MI455X (gfx1250) — matrix operand order repaired and checked by reading, not compiled, not run on hardware
//
#include <hip/hip_runtime.h>
#include <hip/hip_bf16.h>

// ---------------------------------------------------------------------------
// Problem constants (from reference: B=4, N=64, S=4, y_dim=16, g_dim=32, H=256)
// ---------------------------------------------------------------------------
#define MROWS 262144          // B*N*N*S*S
#define HID   256

typedef __attribute__((ext_vector_type(16))) _Float16 v16h;
typedef __attribute__((ext_vector_type(8)))  _Float16 v8h;
typedef __attribute__((ext_vector_type(8)))  float    v8f;
typedef __attribute__((ext_vector_type(4)))  float    v4f;

// IGroupLP scheduling pins: 0x008 = MFMA/WMMA class, 0x100 = DS read class
#define SCHED_WMMA(n) __builtin_amdgcn_sched_group_barrier(0x008, (n), 0)
#define SCHED_DSR(n)  __builtin_amdgcn_sched_group_barrier(0x100, (n), 0)

// ---------------------------------------------------------------------------
// Weight pre-swizzle into WMMA B-fragment layout (f16).
// B 32x16 (KxN) fragment, wave32:  lane L -> col N = L&15,
//   element e (0..15) -> K = (e>=8 ? 16 : 0) + (L>=16 ? 8 : 0) + (e&7)   (2 halves per VGPR, low first)
// Fragment f = kf*16 + nf, stored contiguously: frag*512 + lane*16 + e halves.
// ---------------------------------------------------------------------------
__global__ void prep_wfrag_kernel(const float* __restrict__ W,
                                  _Float16* __restrict__ out, int fi)
{
    int idx = blockIdx.x * 256 + threadIdx.x;
    if (idx >= fi * 256) return;
    int frag = idx >> 9;
    int eif  = idx & 511;
    int lane = eif >> 4;
    int e    = eif & 15;
    int kf   = frag >> 4;
    int nf   = frag & 15;
    int col  = nf * 16 + (lane & 15);
    int k    = kf * 32 + ((e >> 3) << 4) + ((lane >= 16) ? 8 : 0) + (e & 7);
    out[idx] = (_Float16)W[k * 256 + col];
}

// ---------------------------------------------------------------------------
// k_y layer-1 factorization: p1 = cf @ W1[:16], p2 = cf @ W1[16:]
// cf: 1024 x 16 (B*N*S rows), W1: 32 x 256. One block per row, thread per col.
// ---------------------------------------------------------------------------
__global__ void __launch_bounds__(256)
ky_p_kernel(const float* __restrict__ cf, const float* __restrict__ W1,
            float* __restrict__ p1, float* __restrict__ p2)
{
    __shared__ float s_cf[16];
    int row = blockIdx.x, c = threadIdx.x;
    if (c < 16) s_cf[c] = cf[row * 16 + c];
    __syncthreads();
    float a1 = 0.f, a2 = 0.f;
#pragma unroll
    for (int k = 0; k < 16; ++k) {
        a1 += s_cf[k] * W1[k * 256 + c];
        a2 += s_cf[k] * W1[(16 + k) * 256 + c];
    }
    p1[row * 256 + c] = a1;
    p2[row * 256 + c] = a2;
}

// ---------------------------------------------------------------------------
// k_y layer-1 broadcast add: Y[r,:] = p1[b,j,t,:] + p2[b,i,s,:] + b1
// Store f16 + accumulate per-channel sum/sumsq. 8 rows per block.
// ---------------------------------------------------------------------------
__global__ void __launch_bounds__(256)
ky_broadcast_kernel(const float* __restrict__ p1, const float* __restrict__ p2,
                    const float* __restrict__ b1,
                    _Float16* __restrict__ Yout, float* __restrict__ gstats)
{
    __shared__ float s_sum[256], s_ss[256];
    int tid = threadIdx.x;
    s_sum[tid] = 0.f; s_ss[tid] = 0.f;
    __syncthreads();

    int r    = blockIdx.x * 8 + (tid >> 5);
    int lane = tid & 31;
    int c0   = lane * 8;
    // r = ((((b*64 + i)*64 + j)*4 + s)*4 + t)
    int tt = r & 3, ss = (r >> 2) & 3, jj = (r >> 4) & 63, ii = (r >> 10) & 63, bb = (r >> 16) & 3;
    const float* p1r = p1 + (size_t)(((bb * 64 + jj) * 4 + tt) * 256 + c0);
    const float* p2r = p2 + (size_t)(((bb * 64 + ii) * 4 + ss) * 256 + c0);
    _Float16* yr = Yout + (size_t)r * 256 + c0;
#pragma unroll
    for (int e = 0; e < 8; ++e) {
        float v = p1r[e] + p2r[e] + b1[c0 + e];
        yr[e] = (_Float16)v;
        atomicAdd(&s_sum[c0 + e], v);
        atomicAdd(&s_ss[c0 + e], v * v);
    }
    __syncthreads();
    atomicAdd(&gstats[tid], s_sum[tid]);
    atomicAdd(&gstats[256 + tid], s_ss[tid]);
}

// ---------------------------------------------------------------------------
// BN fold: scale = g * rsqrt(var+eps), shift = be - mu*scale
// ---------------------------------------------------------------------------
__global__ void finalize_kernel(const float* __restrict__ stats,
                                const float* __restrict__ gamma,
                                const float* __restrict__ beta,
                                float invCount,
                                float* __restrict__ outScale,
                                float* __restrict__ outShift)
{
    int c = threadIdx.x;
    float mu  = stats[c] * invCount;
    float var = stats[256 + c] * invCount - mu * mu;
    float sc  = gamma[c] * rsqrtf(var + 1e-5f);
    outScale[c] = sc;
    outShift[c] = beta[c] - mu * sc;
}

// ---------------------------------------------------------------------------
// Shared epilogue: add bias, accumulate stats, store f16 tile.
// C/D layout: lane L, vgpr r -> row = rowBase + r + (L>=16?8:0), col = nf*16 + (L&15)
// Each b16 store instruction covers two fully-used 32B segments (lanes 0-15 and
// 16-31 write contiguous 32B runs of the same two rows).
// ---------------------------------------------------------------------------
__device__ __forceinline__ void tile_epilogue(const v8f acc[16], int lane, int rowBase,
                                              const float* s_bias, float* s_sum, float* s_ss,
                                              _Float16* __restrict__ Yout)
{
    int halfsel = (lane >= 16) ? 8 : 0;
    int col0 = lane & 15;
#pragma unroll
    for (int nf = 0; nf < 16; ++nf) {
        int col = nf * 16 + col0;
        float bias = s_bias[col];
        float ls = 0.f, lss = 0.f;
#pragma unroll
        for (int r = 0; r < 8; ++r) {
            float v = acc[nf][r] + bias;
            ls += v; lss += v * v;
            Yout[(size_t)(rowBase + halfsel + r) * 256 + col] = (_Float16)v;
        }
        atomicAdd(&s_sum[col], ls);
        atomicAdd(&s_ss[col], lss);
    }
}

// ---------------------------------------------------------------------------
// Fused BN(scale/shift) + ReLU + GEMM (Mx256)x(256x256) via v_wmma_f32_16x16x32_f16.
// Block = 8 waves, wave computes 16 rows x 256 cols (128 WMMAs).
// LDS holds only the 128KB weight fragments (+bias/stats); BN scale/shift are
// read from global (VMEM class) so the DS-read pipeline contains exactly the
// B fragments. sched_group_barrier pins the interleave  DS(8 preload) then
// [WMMA(1), DS_READ(2)] x 128, giving each fragment a 4-WMMA load-to-use
// distance with ~6 ds_load_b128 in flight (instead of wait dscnt 0x0).
// ---------------------------------------------------------------------------
__global__ void __launch_bounds__(256)
gemm256_bn_kernel(const _Float16* __restrict__ Yin,
                  const float* __restrict__ scale, const float* __restrict__ shift,
                  const _Float16* __restrict__ Wfrag,
                  const float* __restrict__ bias,
                  _Float16* __restrict__ Yout,
                  float* __restrict__ gstats)
{
    extern __shared__ char smem[];
    _Float16* sW     = (_Float16*)smem;            // 65536 halves = 128KB
    float* s_bias    = (float*)(smem + 131072);
    float* s_sum     = s_bias + 256;
    float* s_ss      = s_sum + 256;

    int tid = threadIdx.x;
    {   // cooperative weight-fragment load: 8192 x uint4
        const uint4* src = (const uint4*)Wfrag;
        uint4* dst = (uint4*)sW;
#pragma unroll
        for (int i = 0; i < 32; ++i) dst[tid + 256 * i] = src[tid + 256 * i];
    }
    s_bias[tid]  = bias[tid];
    s_sum[tid] = 0.f; s_ss[tid] = 0.f;
    __syncthreads();

    int wave = tid >> 5, lane = tid & 31;
    int rowBase = blockIdx.x * 128 + wave * 16;
    int myRow = rowBase + (lane & 15);
    int o = (lane < 16) ? 0 : 8;     // A 16x32 f16 layout K offset per lane half
    const _Float16* yrow = Yin + (size_t)myRow * 256;
    const _Float16* sWl = sW + lane * 16;

    v8f acc[16];
    v8f zero = {0.f,0.f,0.f,0.f,0.f,0.f,0.f,0.f};
#pragma unroll
    for (int n = 0; n < 16; ++n) acc[n] = zero;

    // prefetch A for kb=0 and first four B fragments
    v8h ylo = *(const v8h*)(yrow + o);
    v8h yhi = *(const v8h*)(yrow + 16 + o);
    v16h buf[4];
#pragma unroll
    for (int i = 0; i < 4; ++i)
        buf[i] = *(const v16h*)(sWl + i * 512);
    SCHED_DSR(8);   // pin the 4-fragment preload ahead of the WMMA pipeline

#pragma unroll
    for (int kb = 0; kb < 8; ++kb) {
        int k0 = kb * 32 + o;        // elements 0..7  -> K = k0+e
        int k1 = kb * 32 + 16 + o;   // elements 8..15 -> K = k1+e
        // BN scale/shift from global (VMEM, L2-hot) - keeps DS groups exact
        v4f sc0a = *(const v4f*)(scale + k0);
        v4f sc0b = *(const v4f*)(scale + k0 + 4);
        v4f sh0a = *(const v4f*)(shift + k0);
        v4f sh0b = *(const v4f*)(shift + k0 + 4);
        v4f sc1a = *(const v4f*)(scale + k1);
        v4f sc1b = *(const v4f*)(scale + k1 + 4);
        v4f sh1a = *(const v4f*)(shift + k1);
        v4f sh1b = *(const v4f*)(shift + k1 + 4);
        v16h afrag;
#pragma unroll
        for (int e = 0; e < 4; ++e) {
            float f0 = (float)ylo[e]     * sc0a[e] + sh0a[e];
            float f1 = (float)ylo[e + 4] * sc0b[e] + sh0b[e];
            float f2 = (float)yhi[e]     * sc1a[e] + sh1a[e];
            float f3 = (float)yhi[e + 4] * sc1b[e] + sh1b[e];
            afrag[e]      = (_Float16)fmaxf(f0, 0.f);
            afrag[e + 4]  = (_Float16)fmaxf(f1, 0.f);
            afrag[e + 8]  = (_Float16)fmaxf(f2, 0.f);
            afrag[e + 12] = (_Float16)fmaxf(f3, 0.f);
        }
        if (kb < 7) {   // prefetch next K-step's activations under the WMMAs
            ylo = *(const v8h*)(yrow + k0 + 32);
            yhi = *(const v8h*)(yrow + k1 + 32);
        }
#pragma unroll
        for (int nf = 0; nf < 16; ++nf) {
            int lin = kb * 16 + nf;              // linear fragment index 0..127
            acc[nf] = __builtin_amdgcn_wmma_f32_16x16x32_f16(
                false, afrag, false, buf[lin & 3], (short)0, acc[nf], false, false);
            if (lin + 4 < 128)
                buf[lin & 3] = *(const v16h*)(sWl + (lin + 4) * 512);
            SCHED_WMMA(1);
            SCHED_DSR(2);
        }
    }

    tile_epilogue(acc, lane, rowBase, s_bias, s_sum, s_ss, Yout);
    __syncthreads();
    atomicAdd(&gstats[tid], s_sum[tid]);
    atomicAdd(&gstats[256 + tid], s_ss[tid]);
}

// ---------------------------------------------------------------------------
// k_g layer 1: (M x 32 fp32) x (32 x 256) GEMM, single WMMA K-step per tile.
// Same 4-deep round-robin B pipeline with pinned scheduling.
// ---------------------------------------------------------------------------
__global__ void __launch_bounds__(256)
gemm32_kernel(const float* __restrict__ X,
              const _Float16* __restrict__ Wfrag,
              const float* __restrict__ bias,
              _Float16* __restrict__ Yout,
              float* __restrict__ gstats)
{
    __shared__ _Float16 sW[16 * 512];   // 16KB
    __shared__ float s_bias[256], s_sum[256], s_ss[256];
    int tid = threadIdx.x;
    {
        const uint4* src = (const uint4*)Wfrag;
        uint4* dst = (uint4*)sW;
#pragma unroll
        for (int i = 0; i < 4; ++i) dst[tid + 256 * i] = src[tid + 256 * i];
    }
    s_bias[tid] = bias[tid];
    s_sum[tid] = 0.f; s_ss[tid] = 0.f;
    __syncthreads();

    int wave = tid >> 5, lane = tid & 31;
    int rowBase = blockIdx.x * 128 + wave * 16;
    int myRow = rowBase + (lane & 15);
    int o = (lane < 16) ? 0 : 8;
    const float* xr = X + (size_t)myRow * 32;
    const _Float16* sWl = sW + lane * 16;

    v4f x0 = ((const v4f*)(xr + o))[0];
    v4f x1 = ((const v4f*)(xr + o))[1];
    v4f x2 = ((const v4f*)(xr + 16 + o))[0];
    v4f x3 = ((const v4f*)(xr + 16 + o))[1];
    v16h afrag;
#pragma unroll
    for (int e = 0; e < 4; ++e) {
        afrag[e]      = (_Float16)x0[e];
        afrag[e + 4]  = (_Float16)x1[e];
        afrag[e + 8]  = (_Float16)x2[e];
        afrag[e + 12] = (_Float16)x3[e];
    }

    v8f acc[16];
    v8f zero = {0.f,0.f,0.f,0.f,0.f,0.f,0.f,0.f};
    v16h buf[4];
#pragma unroll
    for (int i = 0; i < 4; ++i)
        buf[i] = *(const v16h*)(sWl + i * 512);
    SCHED_DSR(8);
#pragma unroll
    for (int nf = 0; nf < 16; ++nf) {
        acc[nf] = __builtin_amdgcn_wmma_f32_16x16x32_f16(
            false, afrag, false, buf[nf & 3], (short)0, zero, false, false);
        if (nf + 4 < 16)
            buf[nf & 3] = *(const v16h*)(sWl + (nf + 4) * 512);
        SCHED_WMMA(1);
        SCHED_DSR(2);
    }

    tile_epilogue(acc, lane, rowBase, s_bias, s_sum, s_ss, Yout);
    __syncthreads();
    atomicAdd(&gstats[tid], s_sum[tid]);
    atomicAdd(&gstats[256 + tid], s_ss[tid]);
}

// ---------------------------------------------------------------------------
// Layer 3: BN+ReLU then dot with W3 (256 -> 1) + b3; scalar output stats.
// One wave per row; 8 rows per block.
// ---------------------------------------------------------------------------
__global__ void __launch_bounds__(256)
layer3_kernel(const _Float16* __restrict__ Yin,
              const float* __restrict__ scale, const float* __restrict__ shift,
              const float* __restrict__ W3, const float* __restrict__ b3,
              float* __restrict__ z, float* __restrict__ gstats)
{
    __shared__ float s_scale[256], s_shift[256], s_w3[256];
    __shared__ float s_zstat[2];
    int tid = threadIdx.x;
    s_scale[tid] = scale[tid];
    s_shift[tid] = shift[tid];
    s_w3[tid]    = W3[tid];
    if (tid < 2) s_zstat[tid] = 0.f;
    __syncthreads();

    int wave = tid >> 5, lane = tid & 31;
    int row = blockIdx.x * 8 + wave;
    int c0 = lane * 8;
    v8h y = *(const v8h*)(Yin + (size_t)row * 256 + c0);
    float p = 0.f;
#pragma unroll
    for (int e = 0; e < 8; ++e) {
        float v = (float)y[e] * s_scale[c0 + e] + s_shift[c0 + e];
        v = fmaxf(v, 0.f);
        p += v * s_w3[c0 + e];
    }
#pragma unroll
    for (int m = 16; m >= 1; m >>= 1) p += __shfl_xor(p, m, 32);
    if (lane == 0) {
        float zz = p + b3[0];
        z[row] = zz;
        atomicAdd(&s_zstat[0], zz);
        atomicAdd(&s_zstat[1], zz * zz);
    }
    __syncthreads();
    if (tid == 0) {
        atomicAdd(&gstats[0], s_zstat[0]);
        atomicAdd(&gstats[256], s_zstat[1]);
    }
}

// ---------------------------------------------------------------------------
// Final combine: out = relu(bn(z_ky)) + relu(bn(z_kg)); 1-channel BN inline.
// ---------------------------------------------------------------------------
__global__ void __launch_bounds__(256)
combine_kernel(const float* __restrict__ zky, const float* __restrict__ sky,
               const float* __restrict__ g3y, const float* __restrict__ be3y,
               const float* __restrict__ zkg, const float* __restrict__ skg,
               const float* __restrict__ g3g, const float* __restrict__ be3g,
               float* __restrict__ out, float invCount)
{
    int i = blockIdx.x * 256 + threadIdx.x;
    float muY = sky[0] * invCount;
    float vrY = sky[256] * invCount - muY * muY;
    float scY = g3y[0] * rsqrtf(vrY + 1e-5f);
    float shY = be3y[0] - muY * scY;
    float muG = skg[0] * invCount;
    float vrG = skg[256] * invCount - muG * muG;
    float scG = g3g[0] * rsqrtf(vrG + 1e-5f);
    float shG = be3g[0] - muG * scG;
    out[i] = fmaxf(zky[i] * scY + shY, 0.f) + fmaxf(zkg[i] * scG + shG, 0.f);
}

// ---------------------------------------------------------------------------
// Host launcher
// ---------------------------------------------------------------------------
extern "C" void kernel_launch(void* const* d_in, const int* in_sizes, int n_in,
                              void* d_out, int out_size, void* d_ws, size_t ws_size,
                              hipStream_t stream)
{
    (void)in_sizes; (void)n_in; (void)out_size; (void)ws_size;
    const int M = MROWS;
    const float invM = 1.0f / (float)M;

    const float* pg     = (const float*)d_in[0];
    const float* cf     = (const float*)d_in[1];
    // d_in[2] = mask (all ones, unused by reference math)
    const float* ky_W1  = (const float*)d_in[3];
    const float* ky_b1  = (const float*)d_in[4];
    const float* ky_g1  = (const float*)d_in[5];
    const float* ky_be1 = (const float*)d_in[6];
    const float* ky_W2  = (const float*)d_in[7];
    const float* ky_b2  = (const float*)d_in[8];
    const float* ky_g2  = (const float*)d_in[9];
    const float* ky_be2 = (const float*)d_in[10];
    const float* ky_W3  = (const float*)d_in[11];
    const float* ky_b3  = (const float*)d_in[12];
    const float* ky_g3  = (const float*)d_in[13];
    const float* ky_be3 = (const float*)d_in[14];
    const float* kg_W1  = (const float*)d_in[15];
    const float* kg_b1  = (const float*)d_in[16];
    const float* kg_g1  = (const float*)d_in[17];
    const float* kg_be1 = (const float*)d_in[18];
    const float* kg_W2  = (const float*)d_in[19];
    const float* kg_b2  = (const float*)d_in[20];
    const float* kg_g2  = (const float*)d_in[21];
    const float* kg_be2 = (const float*)d_in[22];
    const float* kg_W3  = (const float*)d_in[23];
    const float* kg_b3  = (const float*)d_in[24];
    const float* kg_g3  = (const float*)d_in[25];
    const float* kg_be3 = (const float*)d_in[26];

    char* ws = (char*)d_ws;
    size_t off = 0;
    _Float16* Ya   = (_Float16*)(ws + off); off += (size_t)M * 256 * 2;   // 128MB
    _Float16* Yb   = (_Float16*)(ws + off); off += (size_t)M * 256 * 2;   // 128MB
    float* zky     = (float*)(ws + off);    off += (size_t)M * 4;
    float* zkg     = (float*)(ws + off);    off += (size_t)M * 4;
    float* p1      = (float*)(ws + off);    off += 1024 * 256 * 4;
    float* p2      = (float*)(ws + off);    off += 1024 * 256 * 4;
    _Float16* wfKy2 = (_Float16*)(ws + off); off += 256 * 256 * 2;
    _Float16* wfKg2 = (_Float16*)(ws + off); off += 256 * 256 * 2;
    _Float16* wfKg1 = (_Float16*)(ws + off); off += 32 * 256 * 2;
    float* stats   = (float*)(ws + off);    off += 6 * 512 * 4;           // 6 slots
    float* scsh    = (float*)(ws + off);    off += 4 * 512 * 4;           // 4 scale/shift pairs

    const int GEMM_SMEM = 131072 + 3 * 1024;  // W frags + bias/sum/ss

    hipMemsetAsync(stats, 0, 6 * 512 * sizeof(float), stream);

    // weight swizzles
    prep_wfrag_kernel<<<256, 256, 0, stream>>>(ky_W2, wfKy2, 256);
    prep_wfrag_kernel<<<256, 256, 0, stream>>>(kg_W2, wfKg2, 256);
    prep_wfrag_kernel<<<32, 256, 0, stream>>>(kg_W1, wfKg1, 32);

    // ---- k_y branch ----
    ky_p_kernel<<<1024, 256, 0, stream>>>(cf, ky_W1, p1, p2);
    ky_broadcast_kernel<<<M / 8, 256, 0, stream>>>(p1, p2, ky_b1, Ya, stats + 0 * 512);
    finalize_kernel<<<1, 256, 0, stream>>>(stats + 0 * 512, ky_g1, ky_be1, invM,
                                           scsh + 0 * 512, scsh + 0 * 512 + 256);
    gemm256_bn_kernel<<<M / 128, 256, GEMM_SMEM, stream>>>(
        Ya, scsh + 0 * 512, scsh + 0 * 512 + 256, wfKy2, ky_b2, Yb, stats + 1 * 512);
    finalize_kernel<<<1, 256, 0, stream>>>(stats + 1 * 512, ky_g2, ky_be2, invM,
                                           scsh + 1 * 512, scsh + 1 * 512 + 256);
    layer3_kernel<<<M / 8, 256, 0, stream>>>(
        Yb, scsh + 1 * 512, scsh + 1 * 512 + 256, ky_W3, ky_b3, zky, stats + 2 * 512);

    // ---- k_g branch ----
    gemm32_kernel<<<M / 128, 256, 0, stream>>>(pg, wfKg1, kg_b1, Ya, stats + 3 * 512);
    finalize_kernel<<<1, 256, 0, stream>>>(stats + 3 * 512, kg_g1, kg_be1, invM,
                                           scsh + 2 * 512, scsh + 2 * 512 + 256);
    gemm256_bn_kernel<<<M / 128, 256, GEMM_SMEM, stream>>>(
        Ya, scsh + 2 * 512, scsh + 2 * 512 + 256, wfKg2, kg_b2, Yb, stats + 4 * 512);
    finalize_kernel<<<1, 256, 0, stream>>>(stats + 4 * 512, kg_g2, kg_be2, invM,
                                           scsh + 3 * 512, scsh + 3 * 512 + 256);
    layer3_kernel<<<M / 8, 256, 0, stream>>>(
        Yb, scsh + 3 * 512, scsh + 3 * 512 + 256, kg_W3, kg_b3, zkg, stats + 5 * 512);

    // ---- combine ----
    combine_kernel<<<M / 256, 256, 0, stream>>>(
        zky, stats + 2 * 512, ky_g3, ky_be3,
        zkg, stats + 5 * 512, kg_g3, kg_be3,
        (float*)d_out, invM);
}
